// OPTAttention_77816217469067
// MI455X (gfx1250) — matrix operand order repaired and checked by reading, not compiled, not run on hardware
//
#include <hip/hip_runtime.h>
#include <hip/hip_bf16.h>

typedef __attribute__((ext_vector_type(16))) _Float16 v16h;
typedef __attribute__((ext_vector_type(8)))  _Float16 v8h;
typedef __attribute__((ext_vector_type(4)))  _Float16 v4h;
typedef __attribute__((ext_vector_type(8)))  float    v8f;

union U16h { v16h v; v8h h[2]; };

#define EMBED 2048
#define TTOK  8192
#define HEADS 32
#define HDIM  64
#define SEQ   1024

// ---------------------------------------------------------------------------
// f32 -> f16 conversion (vectorized grid-stride)
// ---------------------------------------------------------------------------
__global__ void cvt_f32_f16(const float4* __restrict__ in,
                            _Float16* __restrict__ out, int n4) {
  int i = blockIdx.x * blockDim.x + threadIdx.x;
  int stride = gridDim.x * blockDim.x;
  for (; i < n4; i += stride) {
    float4 v = in[i];
    v4h o;
    o[0] = (_Float16)v.x; o[1] = (_Float16)v.y;
    o[2] = (_Float16)v.z; o[3] = (_Float16)v.w;
    *(v4h*)(out + (size_t)i * 4) = o;
  }
}

// ---------------------------------------------------------------------------
// WMMA GEMM: out = A[M,K] @ W[N,K]^T + bias[N]
// mode 0: outH = f16 [M,N]        (Q, K projections)
// mode 1: outH = f16 [N,M] (transposed, for V)
// mode 2: outF = f32 [M,N]        (final O projection -> d_out)
// 8 waves per block; each wave computes a 64x32 tile with 8 WMMA accum frags.
// ---------------------------------------------------------------------------
__global__ __launch_bounds__(256) void gemm_f16(
    const _Float16* __restrict__ A,
    const _Float16* __restrict__ Bw,
    const float*    __restrict__ bias,
    _Float16* __restrict__ outH,
    float*    __restrict__ outF,
    int M, int N, int K, int mode) {
  const int lane = threadIdx.x & 31;
  const int wave = threadIdx.x >> 5;
  const int lm = lane & 15;
  const int lh = lane >> 4;

  const int ntiles = N / 32;
  const int wid = blockIdx.x * 8 + wave;
  const int m0 = (wid / ntiles) * 64;
  const int n0 = (wid % ntiles) * 32;

  v8f acc[8] = {};

  const _Float16* arow[4];
  for (int ai = 0; ai < 4; ++ai)
    arow[ai] = A + (size_t)(m0 + 16 * ai + lm) * K;
  const _Float16* brow[2];
  for (int bi = 0; bi < 2; ++bi)
    brow[bi] = Bw + (size_t)(n0 + 16 * bi + lm) * K;

  for (int k = 0; k < K; k += 32) {
    U16h a[4], b[2];
    // A frag: lane lm = M row; 16 contiguous K halves starting at k+lh*16
    for (int ai = 0; ai < 4; ++ai) {
      const _Float16* p = arow[ai] + k + lh * 16;
      a[ai].h[0] = *(const v8h*)(p);
      a[ai].h[1] = *(const v8h*)(p + 8);
    }
    // B frag: lane lm = N col; 16 contiguous K halves starting at k+lh*16
    for (int bi = 0; bi < 2; ++bi) {
      const _Float16* p = brow[bi] + k + lh * 16;
      b[bi].h[0] = *(const v8h*)(p);
      b[bi].h[1] = *(const v8h*)(p + 8);
    }
    for (int ai = 0; ai < 4; ++ai)
      for (int bi = 0; bi < 2; ++bi)
        acc[ai * 2 + bi] = __builtin_amdgcn_wmma_f32_16x16x32_f16(
            false, a[ai].v, false, b[bi].v, (short)0, acc[ai * 2 + bi],
            false, false);
  }

  float bv0 = bias[n0 + lm];
  float bv1 = bias[n0 + 16 + lm];

  for (int ai = 0; ai < 4; ++ai) {
    for (int bi = 0; bi < 2; ++bi) {
      v8f r = acc[ai * 2 + bi];
      float bb = bi ? bv1 : bv0;
      int n = n0 + 16 * bi + lm;
      int mbase = m0 + 16 * ai + 8 * lh;  // C frag: VGPR j -> row j + 8*lh
      if (mode == 0) {
        for (int j = 0; j < 8; ++j)
          outH[(size_t)(mbase + j) * N + n] = (_Float16)(r[j] + bb);
      } else if (mode == 1) {
        v8h pk;
        for (int j = 0; j < 8; ++j) pk[j] = (_Float16)(r[j] + bb);
        *(v8h*)(outH + (size_t)n * M + mbase) = pk;  // contiguous along M
      } else {
        for (int j = 0; j < 8; ++j)
          outF[(size_t)(mbase + j) * N + n] = r[j] + bb;
      }
    }
  }
}

// ---------------------------------------------------------------------------
// Causal attention for one (seq b, head h, 16-row q block).
// Q,K: f16 [T,E] ; Vt: f16 [E,T] ; Ctx: f16 [T,E]
// LDS: S[16][1024] f32 scores/probs (64KB) + reduction scratch.
// ---------------------------------------------------------------------------
__global__ __launch_bounds__(256) void attn_kernel(
    const _Float16* __restrict__ Q,
    const _Float16* __restrict__ Km,
    const _Float16* __restrict__ Vt,
    _Float16* __restrict__ Ctx) {
  extern __shared__ char smem[];
  float* S      = (float*)smem;                          // 16*1024 f32
  float* xred   = (float*)(smem + 65536);                // 4*16*16 f32
  float* red    = (float*)(smem + 65536 + 4096);         // 16*16 f32
  float* rowsum = (float*)(smem + 65536 + 4096 + 1024);  // 16 f32

  const int blk = blockIdx.x;
  const int qb = blk & 63;
  const int h  = (blk >> 6) & 31;
  const int b  = blk >> 11;
  const int lane = threadIdx.x & 31;
  const int wave = threadIdx.x >> 5;
  const int lm = lane & 15;
  const int lh = lane >> 4;
  const int nkt = qb + 1;  // # of valid 16-wide k tiles (causal)
  const size_t qrow0 = (size_t)b * SEQ + (size_t)qb * 16;

  // ---- stage 1: S = 0.125 * Q @ K^T for k tiles [0, nkt)
  U16h aq[2];
  {
    const _Float16* qp = Q + (qrow0 + lm) * EMBED + h * HDIM;
    for (int d = 0; d < 2; ++d) {
      const _Float16* p = qp + d * 32 + lh * 16;
      aq[d].h[0] = *(const v8h*)(p);
      aq[d].h[1] = *(const v8h*)(p + 8);
    }
  }
  for (int kt = wave; kt < nkt; kt += 8) {
    U16h bk[2];
    for (int d = 0; d < 2; ++d) {
      const _Float16* p = Km + ((size_t)b * SEQ + kt * 16 + lm) * EMBED +
                          h * HDIM + d * 32 + lh * 16;
      bk[d].h[0] = *(const v8h*)(p);
      bk[d].h[1] = *(const v8h*)(p + 8);
    }
    v8f s = {};
    s = __builtin_amdgcn_wmma_f32_16x16x32_f16(false, aq[0].v, false, bk[0].v,
                                               (short)0, s, false, false);
    s = __builtin_amdgcn_wmma_f32_16x16x32_f16(false, aq[1].v, false, bk[1].v,
                                               (short)0, s, false, false);
    for (int j = 0; j < 8; ++j)
      S[(j + 8 * lh) * 1024 + kt * 16 + lm] = s[j] * 0.125f;
  }
  __syncthreads();

  // ---- stage 2: masked softmax (unnormalized exp kept in S; sums deferred)
  const int row = threadIdx.x >> 4;
  const int sub = threadIdx.x & 15;
  const int L = qb * 16 + row + 1;  // valid cols: c < L (causal)
  float mx = -3.0e38f;
  for (int j = 0; j < 64; ++j) {
    int c = sub + 16 * j;
    if (c < L) { float v = S[row * 1024 + c]; mx = v > mx ? v : mx; }
  }
  red[row * 16 + sub] = mx;
  __syncthreads();
  if (sub == 0) {
    float m2 = red[row * 16];
    for (int i = 1; i < 16; ++i) {
      float v = red[row * 16 + i];
      m2 = v > m2 ? v : m2;
    }
    red[row * 16] = m2;
  }
  __syncthreads();
  mx = red[row * 16];
  __syncthreads();
  float sum = 0.0f;
  for (int j = 0; j < 64; ++j) {
    int c = sub + 16 * j;
    float p = 0.0f;
    if (c < L) { p = __expf(S[row * 1024 + c] - mx); sum += p; }
    S[row * 1024 + c] = p;  // zero beyond causal limit (covers k padding too)
  }
  red[row * 16 + sub] = sum;
  __syncthreads();
  if (sub == 0) {
    float t = 0.0f;
    for (int i = 0; i < 16; ++i) t += red[row * 16 + i];
    rowsum[row] = t;
  }
  __syncthreads();

  // ---- stage 3: O = P @ V  (A from LDS f32->f16, B from transposed V)
  const int ntile = wave & 3;  // d tile (HDIM/16)
  const int khalf = wave >> 2; // split k range across wave pairs
  const int nsteps = (nkt * 16 + 31) >> 5;
  v8f o = {};
  for (int j = khalf; j < nsteps; j += 2) {
    int cbase = j * 32;
    U16h ap;
    {
      const float* sp = S + lm * 1024 + cbase + lh * 16;
      for (int e = 0; e < 8; ++e) ap.h[0][e] = (_Float16)sp[e];
      const float* sp2 = sp + 8;
      for (int e = 0; e < 8; ++e) ap.h[1][e] = (_Float16)sp2[e];
    }
    U16h bv;
    const _Float16* vp = Vt + (size_t)(h * HDIM + ntile * 16 + lm) * TTOK +
                         (size_t)b * SEQ + cbase + lh * 16;
    bv.h[0] = *(const v8h*)(vp);
    bv.h[1] = *(const v8h*)(vp + 8);
    o = __builtin_amdgcn_wmma_f32_16x16x32_f16(false, ap.v, false, bv.v,
                                               (short)0, o, false, false);
  }
  if (khalf == 1) {
    for (int j = 0; j < 8; ++j)
      xred[ntile * 256 + (j + 8 * lh) * 16 + lm] = o[j];
  }
  __syncthreads();
  if (khalf == 0) {
    for (int j = 0; j < 8; ++j) {
      int m = j + 8 * lh;
      float val = o[j] + xred[ntile * 256 + m * 16 + lm];
      val /= rowsum[m];
      Ctx[(qrow0 + m) * EMBED + h * HDIM + ntile * 16 + lm] = (_Float16)val;
    }
  }
}

// ---------------------------------------------------------------------------
extern "C" void kernel_launch(void* const* d_in, const int* in_sizes, int n_in,
                              void* d_out, int out_size, void* d_ws,
                              size_t ws_size, hipStream_t stream) {
  const float* x  = (const float*)d_in[0];
  const float* Wq = (const float*)d_in[1];
  const float* bq = (const float*)d_in[2];
  const float* Wk = (const float*)d_in[3];
  const float* bk = (const float*)d_in[4];
  const float* Wv = (const float*)d_in[5];
  const float* bv = (const float*)d_in[6];
  const float* Wo = (const float*)d_in[7];
  const float* bo = (const float*)d_in[8];
  float* out = (float*)d_out;

  const size_t NX = (size_t)TTOK * EMBED;     // 16,777,216
  const size_t NW = (size_t)EMBED * EMBED;    // 4,194,304
  _Float16* ws  = (_Float16*)d_ws;
  _Float16* Xh  = ws;
  _Float16* Qh  = ws + NX;
  _Float16* Kh  = ws + 2 * NX;
  _Float16* Vt  = ws + 3 * NX;                // [E, T] transposed
  _Float16* Ctx = ws + 4 * NX;
  _Float16* Wqh = ws + 5 * NX;
  _Float16* Wkh = Wqh + NW;
  _Float16* Wvh = Wkh + NW;
  _Float16* Woh = Wvh + NW;

  cvt_f32_f16<<<2048, 256, 0, stream>>>((const float4*)x,  Xh,  (int)(NX / 4));
  cvt_f32_f16<<<1024, 256, 0, stream>>>((const float4*)Wq, Wqh, (int)(NW / 4));
  cvt_f32_f16<<<1024, 256, 0, stream>>>((const float4*)Wk, Wkh, (int)(NW / 4));
  cvt_f32_f16<<<1024, 256, 0, stream>>>((const float4*)Wv, Wvh, (int)(NW / 4));
  cvt_f32_f16<<<1024, 256, 0, stream>>>((const float4*)Wo, Woh, (int)(NW / 4));

  // (M/64)*(N/32) = 128*64 = 8192 waves -> 1024 blocks of 8 waves
  gemm_f16<<<1024, 256, 0, stream>>>(Xh, Wqh, bq, Qh, nullptr,
                                     TTOK, EMBED, EMBED, 0);
  gemm_f16<<<1024, 256, 0, stream>>>(Xh, Wkh, bk, Kh, nullptr,
                                     TTOK, EMBED, EMBED, 0);
  gemm_f16<<<1024, 256, 0, stream>>>(Xh, Wvh, bv, Vt, nullptr,
                                     TTOK, EMBED, EMBED, 1);

  // 8 seqs * 32 heads * 64 q-blocks = 16384 blocks; 70,720 B dynamic LDS
  attn_kernel<<<16384, 256, 70720, stream>>>(Qh, Kh, Vt, Ctx);

  gemm_f16<<<1024, 256, 0, stream>>>(Ctx, Woh, bo, nullptr, out,
                                     TTOK, EMBED, EMBED, 2);
}
